// AttentionLayer_8615704395879
// MI455X (gfx1250) — matrix operand order repaired and checked by reading, not compiled, not run on hardware
//
#include <hip/hip_runtime.h>
#include <hip/hip_bf16.h>
#include <math.h>

typedef __attribute__((ext_vector_type(16))) __bf16 v16bf;
typedef __attribute__((ext_vector_type(8)))  float  v8f;

union BFrag { v16bf v; uint4 q[2]; };

__device__ __forceinline__ unsigned int f2bf(float f) {
    unsigned int u = __float_as_uint(f);
    u += 0x7FFFu + ((u >> 16) & 1u);   // round-to-nearest-even
    return u >> 16;
}

#define B_   32
#define N_   4096
#define E_   512
#define H_   512
#define MT   32            // rows per block
#define APAD 520           // padded LDS row stride (bf16 elems) -> conflict-free frag reads

// ---------------------------------------------------------------------------
// dbias[b][h] = dec[b] @ W_d[:,h] + b_d[h] + b_e[h]
__global__ __launch_bounds__(512) void dproj_kernel(
    const float* __restrict__ dec, const float* __restrict__ W_d,
    const float* __restrict__ b_d, const float* __restrict__ b_e,
    float* __restrict__ dbias)
{
    const int b = blockIdx.x;
    const int h = threadIdx.x;
    const float* dv = dec + b * 512;
    float acc = b_d[h] + b_e[h];
    for (int d = 0; d < 512; ++d)
        acc += dv[d] * W_d[d * 512 + h];
    dbias[b * 512 + h] = acc;
}

// ---------------------------------------------------------------------------
// Wt[h*512+k] = bf16(W_e[k*512+h])  (transpose + convert; stays in L2)
__global__ __launch_bounds__(256) void mkWt_kernel(
    const float* __restrict__ W_e, unsigned short* __restrict__ Wt)
{
    const int idx = blockIdx.x * 256 + threadIdx.x;
    const int h = idx >> 9, k = idx & 511;
    Wt[idx] = (unsigned short)f2bf(W_e[k * 512 + h]);
}

// ---------------------------------------------------------------------------
// Fused: logits[r] = mask ? (W_o . tanh(enc[r]@W_e + dbias[b]) + b_o) : -1e9
__global__ __launch_bounds__(256) void fused_gemm_kernel(
    const float* __restrict__ enc, const unsigned short* __restrict__ Wt,
    const float* __restrict__ dbias, const float* __restrict__ W_o,
    const float* __restrict__ b_o, const int* __restrict__ mask,
    float* __restrict__ logits)
{
    __shared__ __align__(16) unsigned short Als[MT * APAD];
    __shared__ float lred[MT];

    const int t    = threadIdx.x;
    const int wave = t >> 5;
    const int lane = t & 31;
    const long r0  = (long)blockIdx.x * MT;      // flattened row b*N + n
    const int  b   = (int)(r0 >> 12);            // N = 4096

    if (t < MT) lred[t] = 0.0f;

    // Cooperative coalesced load: 32 rows x 512 fp32 -> bf16 LDS tile
    const float* encBase = enc + r0 * E_;
    #pragma unroll
    for (int i = 0; i < 16; ++i) {
        const int c    = t + i * 256;
        const int flat = c * 4;
        const int row  = flat >> 9;
        const int col  = flat & 511;
        const float4 v = *(const float4*)(encBase + row * 512 + col);
        uint2 p;
        p.x = f2bf(v.x) | (f2bf(v.y) << 16);
        p.y = f2bf(v.z) | (f2bf(v.w) << 16);
        *(uint2*)(&Als[row * APAD + col]) = p;
    }
    __syncthreads();

    const int colh = lane & 15;
    const int hi   = lane >> 4;                  // 0 or 1 (lane half)
    const int aoff0 = colh * APAD + hi * 8;      // row-tile 0, this lane's A row
    const int aoff1 = (16 + colh) * APAD + hi * 8;
    const unsigned short* wrow = Wt + ((wave * 64 + colh) * 512) + hi * 8;

    v8f acc[2][4];
    #pragma unroll
    for (int rt = 0; rt < 2; ++rt)
        #pragma unroll
        for (int ht = 0; ht < 4; ++ht)
            acc[rt][ht] = {};

    // K loop: 512 = 16 steps of 32
    for (int kb = 0; kb < 512; kb += 32) {
        BFrag a0, a1;
        a0.q[0] = *(const uint4*)(&Als[aoff0 + kb]);      // K = kb + 8*hi + 0..7
        a0.q[1] = *(const uint4*)(&Als[aoff0 + kb + 16]); // K = kb+16 + 8*hi + 0..7
        a1.q[0] = *(const uint4*)(&Als[aoff1 + kb]);
        a1.q[1] = *(const uint4*)(&Als[aoff1 + kb + 16]);
        #pragma unroll
        for (int ht = 0; ht < 4; ++ht) {
            BFrag bf;
            const unsigned short* wp = wrow + ht * 16 * 512 + kb;
            bf.q[0] = *(const uint4*)(wp);       // K = kb + 8*hi + 0..7
            bf.q[1] = *(const uint4*)(wp + 16);  // K = kb+16 + 8*hi + 0..7
            acc[0][ht] = __builtin_amdgcn_wmma_f32_16x16x32_bf16(
                false, a0.v, false, bf.v, (short)0, acc[0][ht], false, false);
            acc[1][ht] = __builtin_amdgcn_wmma_f32_16x16x32_bf16(
                false, a1.v, false, bf.v, (short)0, acc[1][ht], false, false);
        }
    }

    // Epilogue: tanh + dot with W_o, reduce over H
    float part[2][8];
    #pragma unroll
    for (int rt = 0; rt < 2; ++rt)
        #pragma unroll
        for (int v = 0; v < 8; ++v)
            part[rt][v] = 0.0f;

    #pragma unroll
    for (int ht = 0; ht < 4; ++ht) {
        const int h = wave * 64 + ht * 16 + colh;
        const float wv = W_o[h];
        const float db = dbias[b * 512 + h];
        #pragma unroll
        for (int v = 0; v < 8; ++v) {
            part[0][v] += tanhf(acc[0][ht][v] + db) * wv;
            part[1][v] += tanhf(acc[1][ht][v] + db) * wv;
        }
    }
    // butterfly over the 16 lane-columns (stays within lane halves)
    #pragma unroll
    for (int m = 1; m < 16; m <<= 1)
        #pragma unroll
        for (int rt = 0; rt < 2; ++rt)
            #pragma unroll
            for (int v = 0; v < 8; ++v)
                part[rt][v] += __shfl_xor(part[rt][v], m, 32);

    if (colh == 0) {   // lanes 0 and 16: rows v+8*hi
        #pragma unroll
        for (int rt = 0; rt < 2; ++rt)
            #pragma unroll
            for (int v = 0; v < 8; ++v)
                atomicAdd(&lred[rt * 16 + hi * 8 + v], part[rt][v]);
    }
    __syncthreads();

    if (t < MT) {
        const long r = r0 + t;
        const int  n = (int)(r & (N_ - 1));
        const float lg = lred[t] + b_o[0];
        logits[r] = (mask[b * N_ + n] != 0) ? lg : -1.0e9f;
    }
}

// ---------------------------------------------------------------------------
// Softmax over N per batch; also zeroes this batch's attn row.
__global__ __launch_bounds__(256) void softmax_kernel(
    const float* __restrict__ logits, float* __restrict__ probs,
    float* __restrict__ attn)
{
    __shared__ float red[256];
    const int b = blockIdx.x, t = threadIdx.x;

    attn[b * 512 + t]       = 0.0f;
    attn[b * 512 + 256 + t] = 0.0f;

    float v[16];
    float mx = -INFINITY;
    #pragma unroll
    for (int i = 0; i < 16; ++i) {
        v[i] = logits[b * N_ + i * 256 + t];
        mx = fmaxf(mx, v[i]);
    }
    red[t] = mx; __syncthreads();
    for (int s = 128; s > 0; s >>= 1) {
        if (t < s) red[t] = fmaxf(red[t], red[t + s]);
        __syncthreads();
    }
    mx = red[0]; __syncthreads();

    float sum = 0.0f;
    #pragma unroll
    for (int i = 0; i < 16; ++i) { v[i] = __expf(v[i] - mx); sum += v[i]; }
    red[t] = sum; __syncthreads();
    for (int s = 128; s > 0; s >>= 1) {
        if (t < s) red[t] += red[t + s];
        __syncthreads();
    }
    const float inv = 1.0f / red[0];
    #pragma unroll
    for (int i = 0; i < 16; ++i)
        probs[b * N_ + i * 256 + t] = v[i] * inv;
}

// ---------------------------------------------------------------------------
// attn[b][e] += sum_n probs[b][n] * enc[b][n][e]   (128 n per block)
__global__ __launch_bounds__(256) void attn_kernel(
    const float* __restrict__ enc, const float* __restrict__ probs,
    float* __restrict__ attn)
{
    __shared__ float p[128];
    const int b = blockIdx.y, chunk = blockIdx.x, t = threadIdx.x;
    const int n0 = chunk * 128;
    if (t < 128) p[t] = probs[b * N_ + n0 + t];
    __syncthreads();

    float a0 = 0.0f, a1 = 0.0f;
    const float* base = enc + ((long)b * N_ + n0) * E_;
    for (int i = 0; i < 128; ++i) {
        const float pi = p[i];
        a0 += pi * base[i * 512 + t];
        a1 += pi * base[i * 512 + 256 + t];
    }
    unsafeAtomicAdd(&attn[b * 512 + t], a0);
    unsafeAtomicAdd(&attn[b * 512 + 256 + t], a1);
}

// ---------------------------------------------------------------------------
extern "C" void kernel_launch(void* const* d_in, const int* in_sizes, int n_in,
                              void* d_out, int out_size, void* d_ws, size_t ws_size,
                              hipStream_t stream) {
    const float* enc  = (const float*)d_in[0];
    const float* dec  = (const float*)d_in[1];
    const int*   mask = (const int*)  d_in[2];
    const float* W_e  = (const float*)d_in[3];
    const float* b_e  = (const float*)d_in[4];
    const float* W_d  = (const float*)d_in[5];
    const float* b_d  = (const float*)d_in[6];
    const float* W_o  = (const float*)d_in[7];
    const float* b_o  = (const float*)d_in[8];

    float* out   = (float*)d_out;
    float* attn  = out;                 // [B, E]   = 16384 floats
    float* probs = out + B_ * E_;       // [B, N]   = 131072 floats

    char* ws = (char*)d_ws;
    float*          dbias  = (float*)ws;                               // 64 KB
    unsigned short* Wt     = (unsigned short*)(ws + (64 << 10));       // 512 KB
    float*          logits = (float*)(ws + (64 << 10) + (512 << 10));  // 512 KB

    dproj_kernel<<<B_, 512, 0, stream>>>(dec, W_d, b_d, b_e, dbias);
    mkWt_kernel<<<(H_ * E_) / 256, 256, 0, stream>>>(W_e, Wt);
    fused_gemm_kernel<<<(B_ * N_) / MT, 256, 0, stream>>>(
        enc, Wt, dbias, W_o, b_o, mask, logits);
    softmax_kernel<<<B_, 256, 0, stream>>>(logits, probs, attn);
    attn_kernel<<<dim3(N_ / 128, B_), 256, 0, stream>>>(enc, probs, attn);
}
